// BioGNN_40243843564118
// MI455X (gfx1250) — matrix operand order repaired and checked by reading, not compiled, not run on hardware
//
#include <hip/hip_runtime.h>
#include <hip/hip_bf16.h>
#include <stdint.h>

// ---------------- problem constants ----------------
#define N_NODES   20000
#define N_EDGES   320000
#define E_TOT     (N_EDGES + N_NODES)   // self loops appended
#define F_IN      512
#define HID       64
#define N_GRAPHS  64
#define N_CLASSES 5
#define SLOPE     0.2f
#define EPS_BN    1e-5f
#define MT_TOTAL  (N_NODES / 16)        // 1250 row tiles (exact)

// TDM availability (device pass on CDNA5 toolchains only; host pass -> 0)
#ifdef __has_builtin
#if __has_builtin(__builtin_amdgcn_tensor_load_to_lds)
#define HAVE_TDM 1
#endif
#endif
#ifndef HAVE_TDM
#define HAVE_TDM 0
#endif

// ---------------- WMMA / TDM types ----------------
typedef __attribute__((ext_vector_type(16))) __bf16 v16bf;
typedef __attribute__((ext_vector_type(8)))  float  v8f;
typedef __attribute__((ext_vector_type(4)))  unsigned int v4u;
typedef __attribute__((ext_vector_type(8)))  int    v8i;
typedef __attribute__((ext_vector_type(4)))  int    v4i;
struct BFrag { uint4 lo, hi; };                      // 32B raw -> v16bf
static_assert(sizeof(BFrag) == sizeof(v16bf), "frag size mismatch");

__device__ __forceinline__ unsigned short f2bf(float f) {   // RNE f32->bf16
  unsigned u = __float_as_uint(f);
  u += 0x7FFFu + ((u >> 16) & 1u);
  return (unsigned short)(u >> 16);
}
// order-preserving float<->uint key for atomicMax-based segment max
__device__ __forceinline__ unsigned fkey(float f) {
  unsigned b = __float_as_uint(f);
  return (b & 0x80000000u) ? ~b : (b | 0x80000000u);
}
__device__ __forceinline__ float funkey(unsigned k) {
  unsigned b = (k & 0x80000000u) ? (k & 0x7FFFFFFFu) : ~k;
  return __uint_as_float(b);
}
__device__ __forceinline__ float lrelu(float x) { return x > 0.f ? x : SLOPE * x; }

// ---------------- elementwise f32 -> bf16 ----------------
__global__ __launch_bounds__(256) void k_f32_to_bf16(const float* __restrict__ src,
                                                     unsigned short* __restrict__ dst, int n) {
  int i = blockIdx.x * 256 + threadIdx.x;
  if (i < n) dst[i] = f2bf(src[i]);
}

// ---------------- prepack W into B-fragment order ----------------
// Bp[((kt*ntiles + nt)*32 + lane)*16 + i] with (per ISA B 32x16 layout)
//   k = kt*32 + (i>>3)*16 + (lane>>4)*8 + (i&7) ,  n = nt*16 + (lane&15)
__global__ __launch_bounds__(256) void k_prepack_b(const float* __restrict__ W,
                                                   unsigned short* __restrict__ Bp,
                                                   int K, int N) {
  int idx = blockIdx.x * 256 + threadIdx.x;
  if (idx >= K * N) return;
  int i    = idx & 15;
  int lane = (idx >> 4) & 31;
  int tile = idx >> 9;
  int ntiles = N >> 4;
  int nt = tile % ntiles;
  int kt = tile / ntiles;
  int k = (kt << 5) + ((i >> 3) << 4) + ((lane >> 4) << 3) + (i & 7);
  int n = (nt << 4) + (lane & 15);
  Bp[idx] = f2bf(W[k * N + n]);
}

// ---------------- bf16 WMMA GEMM: C[M,N] = A[M,K] * B[K,N] ----------------
// block = 256 threads (8 waves); each wave computes a 16x64 output strip,
// all 8 waves share one LDS-staged packed-B strip (rows x 4KB, contiguous),
// loaded by a single TDM tensor_load_to_lds issued by wave 0.
// K is a template constant -> fully unrolled, branch-free, copy-free pipeline.
template <int K>
__global__ __launch_bounds__(256) void k_gemm_bf16(const unsigned short* __restrict__ A,
                                                   const unsigned short* __restrict__ Bp,
                                                   float* __restrict__ C,
                                                   int N) {
  constexpr int ROWS = K >> 5;                      // number of 32-deep k-tiles
  extern __shared__ unsigned short Bs[];            // ROWS * 2048 ushorts
  const int lane   = threadIdx.x & 31;
  const int wave   = threadIdx.x >> 5;
  const int ntiles = N >> 4;
  const int mtile  = blockIdx.x * 8 + wave;
  const int mclamp = mtile < MT_TOTAL ? mtile : (MT_TOTAL - 1);
  const int m0 = mclamp << 4;
  const int n0 = blockIdx.y << 6;

#if HAVE_TDM
  if (wave == 0) {                                  // wave-uniform: EXEC all ones
    // ---- Tensor DMA descriptor (D#), per CDNA5 ISA ----
    unsigned long long ga = (unsigned long long)(Bp + (size_t)(n0 >> 4) * 512);
    unsigned lds_off = (unsigned)(uintptr_t)&Bs[0]; // LDS aperture: low 32 bits = byte offset
    unsigned dim0 = (unsigned)(ntiles * 512);       // elements (2B each) per packed row
    v4u g0;
    g0[0] = 1u;                                     // count=1, user descriptor
    g0[1] = lds_off;                                // lds_addr
    g0[2] = (unsigned)ga;                           // global_addr[31:0]
    g0[3] = (unsigned)(ga >> 32) | 0x80000000u;     // global_addr[56:32] | type=2
    v8i g1;
    g1[0] = (int)(1u << 16);                        // data_size=1 (2 bytes); mask/pad/iter=0
    g1[1] = (int)((dim0 & 0xFFFFu) << 16);          // tensor_dim0[15:0]
    g1[2] = (int)((dim0 >> 16) | ((unsigned)ROWS << 16)); // tensor_dim0 hi | tensor_dim1 lo
    g1[3] = (int)(2048u << 16);                     // tensor_dim1 hi=0 | tile_dim0=2048
    g1[4] = (int)ROWS;                              // tile_dim1=ROWS, tile_dim2=0
    g1[5] = (int)dim0;                              // tensor_dim0_stride[31:0]
    g1[6] = (int)(dim0 >> 16);                      // tensor_dim0_stride[47:32]
    g1[7] = 0;
    v4i z4 = {0, 0, 0, 0};
#if __clang_major__ >= 23
    v8i z8 = {0, 0, 0, 0, 0, 0, 0, 0};
    __builtin_amdgcn_tensor_load_to_lds(g0, g1, z4, z4, z8, 0);
#else
    __builtin_amdgcn_tensor_load_to_lds(g0, g1, z4, z4, 0);
#endif
    __builtin_amdgcn_s_wait_tensorcnt(0);
  }
#else
  // fallback: cooperative global->LDS copy of the packed B strip
  for (int r = 0; r < ROWS; ++r) {
    ((uint4*)(Bs + r * 2048))[threadIdx.x] =
        ((const uint4*)(Bp + ((size_t)r * ntiles + (n0 >> 4)) * 512))[threadIdx.x];
  }
#endif
  __syncthreads();

  v8f acc0 = {}, acc1 = {}, acc2 = {}, acc3 = {};
  const unsigned short* arow = A + (size_t)(m0 + (lane & 15)) * K + ((lane >> 4) << 3);
  const unsigned short* bl   = Bs + lane * 16;

  BFrag ac;                                         // A double buffer: current
  ac.lo = *(const uint4*)(arow);
  ac.hi = *(const uint4*)(arow + 16);
  arow += 32;

#pragma unroll
  for (int kt = 0; kt < ROWS; ++kt) {
    BFrag an = ac;                                  // next (constant-folded per iteration)
    if (kt + 1 < ROWS) {
      an.lo = *(const uint4*)(arow);
      an.hi = *(const uint4*)(arow + 16);
      arow += 32;
      __builtin_prefetch(arow, 0, 1);               // global_prefetch_b8 on A stream
    }
    const unsigned short* bt = bl + kt * 2048;      // this k-tile's 4 n-tiles in LDS
    v16bf af = __builtin_bit_cast(v16bf, ac);
    v16bf b0 = __builtin_bit_cast(v16bf, *(const BFrag*)(bt));
    v16bf b1 = __builtin_bit_cast(v16bf, *(const BFrag*)(bt + 512));
    v16bf b2 = __builtin_bit_cast(v16bf, *(const BFrag*)(bt + 1024));
    v16bf b3 = __builtin_bit_cast(v16bf, *(const BFrag*)(bt + 1536));
    acc0 = __builtin_amdgcn_wmma_f32_16x16x32_bf16(false, af, false, b0, (short)0, acc0, false, false);
    acc1 = __builtin_amdgcn_wmma_f32_16x16x32_bf16(false, af, false, b1, (short)0, acc1, false, false);
    acc2 = __builtin_amdgcn_wmma_f32_16x16x32_bf16(false, af, false, b2, (short)0, acc2, false, false);
    acc3 = __builtin_amdgcn_wmma_f32_16x16x32_bf16(false, af, false, b3, (short)0, acc3, false, false);
    ac = an;
  }

  if (mtile < MT_TOTAL) {
    // C/D layout: VGPR r -> row m0 + r + 8*(lane>>4), col = n0 + (lane&15)
    const int col   = n0 + (lane & 15);
    const int rbase = m0 + ((lane >> 4) << 3);
#pragma unroll
    for (int r = 0; r < 8; ++r) {
      float* crow = C + (size_t)(rbase + r) * N + col;
      crow[0]  = acc0[r];
      crow[16] = acc1[r];
      crow[32] = acc2[r];
      crow[48] = acc3[r];
    }
  }
}

// ---------------- attention logits: al_s/al_d[n,h] = <h[n,h,:], a[h,:]> ----------------
__global__ __launch_bounds__(256) void k_attn_scores(const float* __restrict__ H,
                                                     const float* __restrict__ a_s,
                                                     const float* __restrict__ a_d,
                                                     float* __restrict__ als,
                                                     float* __restrict__ ald,
                                                     int heads) {
  int idx = blockIdx.x * 256 + threadIdx.x;
  if (idx >= N_NODES * heads) return;
  int h = idx % heads;
  int n = idx / heads;
  const float* hp = H + ((size_t)n * heads + h) * HID;
  const float* sp = a_s + h * HID;
  const float* dp = a_d + h * HID;
  float ss = 0.f, sd = 0.f;
#pragma unroll 8
  for (int c = 0; c < HID; ++c) { float v = hp[c]; ss += v * sp[c]; sd += v * dp[c]; }
  als[idx] = ss;
  ald[idx] = sd;
}

__device__ __forceinline__ void edge_endpoints(const int* __restrict__ ei, int e,
                                               int& src, int& dst) {
  if (e < N_EDGES) { src = ei[e]; dst = ei[N_EDGES + e]; }
  else             { src = dst = e - N_EDGES; }           // self loop
}

// ---------------- pass 1: segment max over dst (ordered-uint atomicMax) ----------------
__global__ __launch_bounds__(256) void k_edge_max(const int* __restrict__ ei,
                                                  const float* __restrict__ als,
                                                  const float* __restrict__ ald,
                                                  unsigned* __restrict__ mx,
                                                  int hshift) {
  size_t idx = (size_t)blockIdx.x * 256 + threadIdx.x;
  size_t tot = (size_t)E_TOT << hshift;
  if (idx >= tot) return;
  int e = (int)(idx >> hshift);
  int h = (int)idx & ((1 << hshift) - 1);
  int src, dst; edge_endpoints(ei, e, src, dst);
  float sc = lrelu(als[(src << hshift) + h] + ald[(dst << hshift) + h]);
  atomicMax(&mx[(dst << hshift) + h], fkey(sc));
}

// ---------------- pass 2: ee = exp(e - m[dst]); denom[dst] += ee ----------------
__global__ __launch_bounds__(256) void k_edge_exp(const int* __restrict__ ei,
                                                  const float* __restrict__ als,
                                                  const float* __restrict__ ald,
                                                  const unsigned* __restrict__ mx,
                                                  float* __restrict__ ee_out,
                                                  float* __restrict__ den,
                                                  int hshift) {
  size_t idx = (size_t)blockIdx.x * 256 + threadIdx.x;
  size_t tot = (size_t)E_TOT << hshift;
  if (idx >= tot) return;
  int e = (int)(idx >> hshift);
  int h = (int)idx & ((1 << hshift) - 1);
  int src, dst; edge_endpoints(ei, e, src, dst);
  float sc = lrelu(als[(src << hshift) + h] + ald[(dst << hshift) + h]);
  float m  = funkey(mx[(dst << hshift) + h]);
  float v  = __expf(sc - m);
  ee_out[idx] = v;
  atomicAdd(&den[(dst << hshift) + h], v);
}

// ---------------- pass 3: ACC[dst,:] += alpha * H[src,:]  (coalesced, 1 ch/lane) ----------------
__global__ __launch_bounds__(256) void k_edge_aggr(const int* __restrict__ ei,
                                                   const float* __restrict__ H,
                                                   const float* __restrict__ EE,
                                                   const float* __restrict__ DEN,
                                                   float* __restrict__ ACC,
                                                   int hcshift, int hshift) {
  size_t idx = (size_t)blockIdx.x * 256 + threadIdx.x;
  int e = (int)(idx >> hcshift);
  if (e >= E_TOT) return;
  int j = (int)idx & ((1 << hcshift) - 1);
  int h = j >> 6;                                   // C == 64 always
  int src, dst; edge_endpoints(ei, e, src, dst);
  float alpha = EE[((size_t)e << hshift) + h] / (DEN[(dst << hshift) + h] + 1e-16f);
  atomicAdd(&ACC[((size_t)dst << hcshift) + j], H[((size_t)src << hcshift) + j] * alpha);
}

// ---------------- epilogue layers 1/2: +bias, BN(eval), ReLU, -> bf16 ----------------
__global__ __launch_bounds__(256) void k_epi_bn(const float* __restrict__ ACC,
                                                const float* __restrict__ bias,
                                                const float* __restrict__ gm,
                                                const float* __restrict__ bt,
                                                const float* __restrict__ mu,
                                                const float* __restrict__ var,
                                                unsigned short* __restrict__ outbf) {
  int idx = blockIdx.x * 256 + threadIdx.x;
  if (idx >= N_NODES * 4 * HID) return;
  int j = idx & (4 * HID - 1);
  float v = ACC[idx] + bias[j];
  v = (v - mu[j]) * rsqrtf(var[j] + EPS_BN) * gm[j] + bt[j];
  outbf[idx] = f2bf(fmaxf(v, 0.f));
}

// ---------------- epilogue layer 3: +bias, ReLU (f32 out) ----------------
__global__ __launch_bounds__(256) void k_epi3(const float* __restrict__ ACC,
                                              const float* __restrict__ bias,
                                              float* __restrict__ out) {
  int idx = blockIdx.x * 256 + threadIdx.x;
  if (idx >= N_NODES * HID) return;
  out[idx] = fmaxf(ACC[idx] + bias[idx & (HID - 1)], 0.f);
}

// ---------------- global mean pool (sums + counts) ----------------
__global__ __launch_bounds__(256) void k_pool(const float* __restrict__ h3,
                                              const int* __restrict__ batch,
                                              float* __restrict__ pool,
                                              float* __restrict__ cnt) {
  int idx = blockIdx.x * 256 + threadIdx.x;
  if (idx >= N_NODES * HID) return;
  int n = idx >> 6, j = idx & (HID - 1);
  int b = batch[n];
  atomicAdd(&pool[b * HID + j], h3[idx]);
  if (j == 0) atomicAdd(&cnt[b], 1.f);
}

// ---------------- final tiny MLP: 64x64 -> relu(64x32) -> 64x5 ----------------
__global__ __launch_bounds__(256) void k_mlp(const float* __restrict__ pool,
                                             const float* __restrict__ cnt,
                                             const float* __restrict__ w1,
                                             const float* __restrict__ b1,
                                             const float* __restrict__ w2,
                                             const float* __restrict__ b2,
                                             float* __restrict__ out) {
  __shared__ float g[N_GRAPHS * HID];
  __shared__ float l1[N_GRAPHS * 32];
  int t = threadIdx.x;
  for (int i = t; i < N_GRAPHS * HID; i += 256)
    g[i] = pool[i] / fmaxf(cnt[i >> 6], 1.0f);
  __syncthreads();
  for (int i = t; i < N_GRAPHS * 32; i += 256) {
    int gi = i >> 5, j = i & 31;
    float s = b1[j];
#pragma unroll 8
    for (int c = 0; c < HID; ++c) s += g[gi * HID + c] * w1[c * 32 + j];
    l1[i] = fmaxf(s, 0.f);
  }
  __syncthreads();
  for (int i = t; i < N_GRAPHS * N_CLASSES; i += 256) {
    int gi = i / N_CLASSES, j = i % N_CLASSES;
    float s = b2[j];
#pragma unroll
    for (int c = 0; c < 32; ++c) s += l1[gi * 32 + c] * w2[c * N_CLASSES + j];
    out[i] = s;
  }
}

// ================== host launcher ==================
extern "C" void kernel_launch(void* const* d_in, const int* in_sizes, int n_in,
                              void* d_out, int out_size, void* d_ws, size_t ws_size,
                              hipStream_t stream) {
  const float* x     = (const float*)d_in[0];
  const int*   ei    = (const int*)d_in[1];
  const int*   batch = (const int*)d_in[2];
  const float* W1    = (const float*)d_in[3];
  const float* a1s   = (const float*)d_in[4];
  const float* a1d   = (const float*)d_in[5];
  const float* b1    = (const float*)d_in[6];
  const float* bn1g  = (const float*)d_in[7];
  const float* bn1b  = (const float*)d_in[8];
  const float* bn1m  = (const float*)d_in[9];
  const float* bn1v  = (const float*)d_in[10];
  const float* W2    = (const float*)d_in[11];
  const float* a2s   = (const float*)d_in[12];
  const float* a2d   = (const float*)d_in[13];
  const float* b2    = (const float*)d_in[14];
  const float* bn2g  = (const float*)d_in[15];
  const float* bn2b  = (const float*)d_in[16];
  const float* bn2m  = (const float*)d_in[17];
  const float* bn2v  = (const float*)d_in[18];
  const float* W3    = (const float*)d_in[19];
  const float* a3s   = (const float*)d_in[20];
  const float* a3d   = (const float*)d_in[21];
  const float* b3    = (const float*)d_in[22];
  const float* l1w   = (const float*)d_in[23];
  const float* l1b   = (const float*)d_in[24];
  const float* l2w   = (const float*)d_in[25];
  const float* l2b   = (const float*)d_in[26];
  float* out = (float*)d_out;

  // ---- workspace carve-out (~69 MB) ----
  char* ws = (char*)d_ws;
  size_t off = 0;
  auto carve = [&](size_t bytes) -> void* {
    void* p = ws + off;
    off += (bytes + 255) & ~(size_t)255;
    return p;
  };
  unsigned short* Abf  = (unsigned short*)carve((size_t)N_NODES * F_IN * 2); // bf16 GEMM input
  unsigned short* Wp   = (unsigned short*)carve((size_t)F_IN * 256 * 2);     // packed weights
  float*          Hb   = (float*)carve((size_t)N_NODES * 256 * 4);           // GEMM output h
  float*          ACC  = (float*)carve((size_t)N_NODES * 256 * 4);           // edge aggregation
  float*          ALS  = (float*)carve((size_t)N_NODES * 4 * 4);
  float*          ALD  = (float*)carve((size_t)N_NODES * 4 * 4);
  unsigned*       MAXB = (unsigned*)carve((size_t)N_NODES * 4 * 4);
  float*          DEN  = (float*)carve((size_t)N_NODES * 4 * 4);
  float*          EE   = (float*)carve((size_t)E_TOT * 4 * 4);
  float*          POOL = (float*)carve((size_t)N_GRAPHS * HID * 4);
  float*          CNT  = (float*)carve((size_t)N_GRAPHS * 4);

  auto run_attn = [&](int heads, int hshift, const float* as_, const float* ad_) {
    int HC = heads * HID;
    int nh = N_NODES * heads;
    k_attn_scores<<<(nh + 255) / 256, 256, 0, stream>>>(Hb, as_, ad_, ALS, ALD, heads);
    hipMemsetAsync(MAXB, 0, (size_t)N_NODES * heads * 4, stream);  // key 0 == -inf sentinel
    hipMemsetAsync(DEN,  0, (size_t)N_NODES * heads * 4, stream);
    hipMemsetAsync(ACC,  0, (size_t)N_NODES * HC * 4, stream);
    size_t eh = (size_t)E_TOT * heads;
    k_edge_max<<<(unsigned)((eh + 255) / 256), 256, 0, stream>>>(ei, ALS, ALD, MAXB, hshift);
    k_edge_exp<<<(unsigned)((eh + 255) / 256), 256, 0, stream>>>(ei, ALS, ALD, MAXB, EE, DEN, hshift);
    size_t ehc = (size_t)E_TOT * HC;
    k_edge_aggr<<<(unsigned)((ehc + 255) / 256), 256, 0, stream>>>(ei, Hb, EE, DEN, ACC,
                                                                   hshift + 6, hshift);
  };

  // ---- layer 1: 512 -> 4x64, BN+ReLU ----
  k_f32_to_bf16<<<(N_NODES * F_IN + 255) / 256, 256, 0, stream>>>(x, Abf, N_NODES * F_IN);
  k_prepack_b<<<(F_IN * 256 + 255) / 256, 256, 0, stream>>>(W1, Wp, F_IN, 256);
  k_gemm_bf16<F_IN><<<dim3((MT_TOTAL + 7) / 8, 4), 256, (F_IN / 32) * 4096, stream>>>(Abf, Wp, Hb, 256);
  run_attn(4, 2, a1s, a1d);
  k_epi_bn<<<(N_NODES * 256 + 255) / 256, 256, 0, stream>>>(ACC, b1, bn1g, bn1b, bn1m, bn1v, Abf);

  // ---- layer 2: 256 -> 4x64, BN+ReLU ----
  k_prepack_b<<<(256 * 256 + 255) / 256, 256, 0, stream>>>(W2, Wp, 256, 256);
  k_gemm_bf16<256><<<dim3((MT_TOTAL + 7) / 8, 4), 256, (256 / 32) * 4096, stream>>>(Abf, Wp, Hb, 256);
  run_attn(4, 2, a2s, a2d);
  k_epi_bn<<<(N_NODES * 256 + 255) / 256, 256, 0, stream>>>(ACC, b2, bn2g, bn2b, bn2m, bn2v, Abf);

  // ---- layer 3: 256 -> 1x64, ReLU ----
  k_prepack_b<<<(256 * 64 + 255) / 256, 256, 0, stream>>>(W3, Wp, 256, 64);
  k_gemm_bf16<256><<<dim3((MT_TOTAL + 7) / 8, 1), 256, (256 / 32) * 4096, stream>>>(Abf, Wp, Hb, 64);
  run_attn(1, 0, a3s, a3d);
  k_epi3<<<(N_NODES * HID + 255) / 256, 256, 0, stream>>>(ACC, b3, Hb);

  // ---- mean pool + MLP head ----
  hipMemsetAsync(POOL, 0, (size_t)N_GRAPHS * HID * 4, stream);
  hipMemsetAsync(CNT,  0, (size_t)N_GRAPHS * 4, stream);
  k_pool<<<(N_NODES * HID + 255) / 256, 256, 0, stream>>>(Hb, batch, POOL, CNT);
  k_mlp<<<1, 256, 0, stream>>>(POOL, CNT, l1w, l1b, l2w, l2b, out);
}
